// KernelInduced_66812511256914
// MI455X (gfx1250) — compile-verified, matrix operand order checked by reading, not run on hardware
//
#include <hip/hip_runtime.h>
#include <hip/hip_bf16.h>

typedef __attribute__((ext_vector_type(16))) _Float16 v16h;
typedef __attribute__((ext_vector_type(8)))  float    v8f;

#define NND 5120      // total nodes
#define NFINE 4096
#define WID 32

// --------- WMMA fragment helpers (per CDNA5 ISA 16-bit layouts) -----------

__device__ __forceinline__ int kmapf(int lane, int j) {
  // element j of a lane's v16h maps to K index:
  return (j & 7) + ((lane >> 4) << 3) + ((j >> 3) << 4);
}

__device__ __forceinline__ v16h load_a_frag(const _Float16* H, int ld, int m0, int kbase, int lane) {
  const int m = m0 + (lane & 15);
  const int hi8 = (lane >> 4) << 3;
  const _Float16* row = H + m * ld + kbase;
  v16h a;
#pragma unroll
  for (int j = 0; j < 16; ++j) {
    int k = (j & 7) + hi8 + ((j >> 3) << 4);
    a[j] = row[k];
  }
  return a;
}

__device__ __forceinline__ v8f wmma_f16(v16h a, v16h b, v8f c) {
  return __builtin_amdgcn_wmma_f32_16x16x32_f16(false, a, false, b, (short)0, c, false, false);
}

__device__ __forceinline__ void store_d_relu_f16(v8f c, _Float16* H, int ld, int m0, int n0,
                                                 int lane, const float* bias) {
  const int n = n0 + (lane & 15);
  const int hi = lane >> 4;
  const float bv = bias[n];
#pragma unroll
  for (int r = 0; r < 8; ++r) {
    float v = c[r] + bv;
    v = v > 0.f ? v : 0.f;
    H[(m0 + hi * 8 + r) * ld + n] = (_Float16)v;
  }
}

__device__ __forceinline__ void fatomic_add(float* p, float v) {
  __hip_atomic_fetch_add(p, v, __ATOMIC_RELAXED, __HIP_MEMORY_SCOPE_AGENT);
}

// --------- weight packing: f32 [Kreal x F] -> f16 B-fragment tiles --------

__global__ void pack_w_k(const float* __restrict__ W, _Float16* __restrict__ dst,
                         int Kreal, int Kpad, int F) {
  int idx = blockIdx.x * blockDim.x + threadIdx.x;
  int total = Kpad * F;
  if (idx >= total) return;
  int tile = idx >> 9;                 // 512 elements per 32x16 tile
  int r = idx & 511;
  int lane = r >> 4, j = r & 15;
  int ntn = F >> 4;
  int kt = tile / ntn, nt = tile % ntn;
  int k = (kt << 5) + kmapf(lane, j);
  int n = (nt << 4) + (lane & 15);
  float v = (k < Kreal) ? W[k * F + n] : 0.f;
  dst[idx] = (_Float16)v;
}

// --------- small utility kernels ------------------------------------------

__global__ void zero_f32_k(float* p, int n) {
  int t = blockIdx.x * blockDim.x + threadIdx.x;
  if (t < n) p[t] = 0.f;
}
__global__ void zero_i32_k(int* p, int n) {
  int t = blockIdx.x * blockDim.x + threadIdx.x;
  if (t < n) p[t] = 0;
}
__global__ void count_k(const int* __restrict__ dst, int* cnt, int E) {
  int t = blockIdx.x * blockDim.x + threadIdx.x;
  if (t < E) atomicAdd(&cnt[dst[t]], 1);
}
__global__ void fc_in_k(const float* __restrict__ x, const float* __restrict__ W,
                        const float* __restrict__ b, float* __restrict__ h) {
  int t = blockIdx.x * blockDim.x + threadIdx.x;
  if (t >= NND * WID) return;
  int v = t >> 5, j = t & 31;
  h[t] = x[v] * W[j] + b[j];
}
__global__ void finalize_k(float* __restrict__ h, const float* __restrict__ s,
                           const int* __restrict__ cnt) {
  int t = blockIdx.x * blockDim.x + threadIdx.x;
  if (t >= NND * WID) return;
  float c = (float)cnt[t >> 5];
  if (c < 1.f) c = 1.f;
  float v = h[t] + s[t] / c;
  h[t] = v > 0.f ? v : 0.f;
}

// --------- fused NNConv edge kernel ---------------------------------------
// 64 edges per block, 256 threads (8 wave32).
// MLP: [6(pad32) -> F1 (relu)] (-> [F1 -> F1 (relu)] if THREE) -> [F1 -> 1024]
// Final layer fused with msg[e][o] = sum_i x[src(e)][i] * (We[e][32i+o]),
// scattered with float atomics into sacc[dst*32+o].
template <int F1, bool THREE>
__global__ __launch_bounds__(256) void edge_conv_k(
    const float* __restrict__ ea, const int* __restrict__ srcI, const int* __restrict__ dstI,
    const float* __restrict__ hbuf, float* __restrict__ sacc,
    const _Float16* __restrict__ W1p, const float* __restrict__ b1,
    const _Float16* __restrict__ W2p, const float* __restrict__ b2,
    const _Float16* __restrict__ W3p, const float* __restrict__ b3) {
  extern __shared__ char smem[];
  _Float16* sH0 = (_Float16*)smem;                                   // 64 x 32 (padded attrs)
  _Float16* sH1 = sH0 + 64 * 32;                                     // 64 x F1
  _Float16* sH2 = THREE ? (sH1 + 64 * F1) : sH1;                     // 64 x F1 (if THREE)
  float* sX = (float*)(sH1 + (THREE ? 2 : 1) * 64 * F1);             // 64 x 32 gathered src feats

  const int lane = threadIdx.x & 31;
  const int wave = threadIdx.x >> 5;
  const int e0 = blockIdx.x * 64;

  // phase 0: load edge attrs (K padded 6->32) and gather source features
#pragma unroll
  for (int it = 0; it < 8; ++it) {
    int t = threadIdx.x + it * 256;
    int e = t >> 5, k = t & 31;
    sH0[t] = (k < 6) ? (_Float16)ea[(e0 + e) * 6 + k] : (_Float16)0.0f;
    int sv = srcI[e0 + e];
    sX[t] = hbuf[sv * WID + k];
  }
  __syncthreads();

  // phase 1: H1 = relu([64x32] @ W1[32xF1] + b1), single K-tile
  {
    constexpr int NT = F1 / 16;
    constexpr int TPW = (4 * NT) / 8;
#pragma unroll
    for (int tt = 0; tt < TPW; ++tt) {
      int tile = wave + 8 * tt;
      int mt = tile / NT, nt = tile % NT;
      v16h a = load_a_frag(sH0, 32, mt * 16, 0, lane);
      v16h b = ((const v16h*)W1p)[nt * 32 + lane];
      v8f acc = {};
      acc = wmma_f16(a, b, acc);
      store_d_relu_f16(acc, sH1, F1, mt * 16, nt * 16, lane, b1);
    }
  }
  __syncthreads();

  // phase 2 (3-layer nets): H2 = relu(H1 @ W2[F1xF1] + b2)
  if constexpr (THREE) {
    constexpr int NT = F1 / 16;
    constexpr int KT = F1 / 32;
    constexpr int TPW = (4 * NT) / 8;
#pragma unroll
    for (int tt = 0; tt < TPW; ++tt) {
      int tile = wave + 8 * tt;
      int mt = tile / NT, nt = tile % NT;
      v8f acc = {};
#pragma unroll
      for (int kt = 0; kt < KT; ++kt) {
        v16h a = load_a_frag(sH1, F1, mt * 16, kt * 32, lane);
        v16h b = ((const v16h*)W2p)[(kt * NT + nt) * 32 + lane];
        acc = wmma_f16(a, b, acc);
      }
      store_d_relu_f16(acc, sH2, F1, mt * 16, nt * 16, lane, b2);
    }
    __syncthreads();
  }

  // phase 3: last layer [64 x F1] @ W3[F1 x 1024], fused with message contraction.
  // wave owns mt = wave>>1 (16 edges) and out-channel half p = wave&1.
  // column c = nt*16+n with nt = 2*i + p  ->  c = 32*i + (16*p + n) = 32*i + o.
  {
    constexpr int KT = F1 / 32;
    const _Float16* sHin = THREE ? sH2 : sH1;
    const int mt = wave >> 1;
    const int p = wave & 1;
    const int hi = lane >> 4;

    v16h afr[KT];
#pragma unroll
    for (int kt = 0; kt < KT; ++kt) afr[kt] = load_a_frag(sHin, F1, mt * 16, kt * 32, lane);

    float macc[8];
#pragma unroll
    for (int r = 0; r < 8; ++r) macc[r] = 0.f;

    for (int i = 0; i < 32; ++i) {
      const int nt = 2 * i + p;
      v8f acc = {};
#pragma unroll
      for (int kt = 0; kt < KT; ++kt) {
        v16h b = ((const v16h*)W3p)[(kt * 64 + nt) * 32 + lane];
        acc = wmma_f16(afr[kt], b, acc);
      }
      const float bv = b3[nt * 16 + (lane & 15)];
#pragma unroll
      for (int r = 0; r < 8; ++r) {
        macc[r] += sX[(mt * 16 + hi * 8 + r) * 32 + i] * (acc[r] + bv);
      }
    }

    const int o = p * 16 + (lane & 15);
#pragma unroll
    for (int r = 0; r < 8; ++r) {
      int e = e0 + mt * 16 + hi * 8 + r;
      int d = dstI[e];
      fatomic_add(&sacc[d * WID + o], macc[r]);
    }
  }
}

// --------- fc_out1: relu(h[:4096] @ W[32x256] + b) via WMMA ---------------
__global__ __launch_bounds__(256) void fc_out1_k(const float* __restrict__ h,
                                                 const _Float16* __restrict__ Wp,
                                                 const float* __restrict__ b,
                                                 float* __restrict__ t1) {
  const int lane = threadIdx.x & 31;
  const int wave = threadIdx.x >> 5;
  const int row0 = (blockIdx.x * 8 + wave) * 16;
  const int m = row0 + (lane & 15);
  const int hi = lane >> 4;
  v16h a;
#pragma unroll
  for (int j = 0; j < 16; ++j) a[j] = (_Float16)h[m * WID + kmapf(lane, j)];
#pragma unroll
  for (int nt = 0; nt < 16; ++nt) {
    v16h bf = ((const v16h*)Wp)[nt * 32 + lane];
    v8f acc = {};
    acc = wmma_f16(a, bf, acc);
    const int n = nt * 16 + (lane & 15);
    const float bv = b[n];
#pragma unroll
    for (int r = 0; r < 8; ++r) {
      float v = acc[r] + bv;
      v = v > 0.f ? v : 0.f;
      t1[(row0 + hi * 8 + r) * 256 + n] = v;
    }
  }
}

__global__ void fc_out2_k(const float* __restrict__ t1, const float* __restrict__ W,
                          const float* __restrict__ b, float* __restrict__ out) {
  int row = blockIdx.x * blockDim.x + threadIdx.x;
  if (row >= NFINE) return;
  float s = 0.f;
#pragma unroll 8
  for (int c = 0; c < 256; ++c) s += t1[row * 256 + c] * W[c];
  out[row] = s + b[0];
}

// --------------------------------------------------------------------------

extern "C" void kernel_launch(void* const* d_in, const int* in_sizes, int n_in,
                              void* d_out, int out_size, void* d_ws, size_t ws_size,
                              hipStream_t stream) {
  (void)in_sizes; (void)n_in; (void)out_size; (void)ws_size;
  // ---- inputs (setup_inputs dict order; params depth-first in file order) ----
  const float* x    = (const float*)d_in[0];
  const float* eam  = (const float*)d_in[1];  // [163840, 6]
  const float* ead  = (const float*)d_in[2];  // [65536, 6]
  const float* eau  = (const float*)d_in[3];  // [65536, 6]
  const int*   eim  = (const int*)d_in[4];    // [2, 163840]
  const int*   eid  = (const int*)d_in[5];    // [2, 65536]
  const int*   eiu  = (const int*)d_in[6];    // [2, 65536]
  const float* fciW = (const float*)d_in[7];   const float* fcib = (const float*)d_in[8];
  const float* c01W1 = (const float*)d_in[9];  const float* c01b1 = (const float*)d_in[10];
  const float* c01W2 = (const float*)d_in[11]; const float* c01b2 = (const float*)d_in[12];
  const float* c11W1 = (const float*)d_in[13]; const float* c11b1 = (const float*)d_in[14];
  const float* c11W2 = (const float*)d_in[15]; const float* c11b2 = (const float*)d_in[16];
  const float* c11W3 = (const float*)d_in[17]; const float* c11b3 = (const float*)d_in[18];
  const float* c10W1 = (const float*)d_in[19]; const float* c10b1 = (const float*)d_in[20];
  const float* c10W2 = (const float*)d_in[21]; const float* c10b2 = (const float*)d_in[22];
  const float* c00W1 = (const float*)d_in[23]; const float* c00b1 = (const float*)d_in[24];
  const float* c00W2 = (const float*)d_in[25]; const float* c00b2 = (const float*)d_in[26];
  const float* c00W3 = (const float*)d_in[27]; const float* c00b3 = (const float*)d_in[28];
  const float* fo1W = (const float*)d_in[29];  const float* fo1b = (const float*)d_in[30];
  const float* fo2W = (const float*)d_in[31];  const float* fo2b = (const float*)d_in[32];
  float* out = (float*)d_out;

  // ---- workspace layout ----
  char* w = (char*)d_ws;
  float* h    = (float*)(w);                       // 5120*32 f32
  float* s    = (float*)(w + 655360);              // 5120*32 f32
  int*   cnt  = (int*)(w + 1310720);               // 4 * 5120 int
  float* t1   = (float*)(w + 1392640);             // 4096*256 f32
  _Float16* wp = (_Float16*)(w + 5586944);
  _Float16* p01_1 = wp;                 // 32*128
  _Float16* p01_2 = p01_1 + 4096;       // 128*1024
  _Float16* p11_1 = p01_2 + 131072;     // 32*128
  _Float16* p11_2 = p11_1 + 4096;       // 128*128
  _Float16* p11_3 = p11_2 + 16384;      // 128*1024
  _Float16* p10_1 = p11_3 + 131072;     // 32*128
  _Float16* p10_2 = p10_1 + 4096;       // 128*1024
  _Float16* p00_1 = p10_2 + 131072;     // 32*256
  _Float16* p00_2 = p00_1 + 8192;       // 256*256
  _Float16* p00_3 = p00_2 + 65536;      // 256*1024
  _Float16* pfc1  = p00_3 + 262144;     // 32*256

  // allow >64KB dynamic LDS for the big fused kernel
  (void)hipFuncSetAttribute((const void*)edge_conv_k<256, true>,
                            hipFuncAttributeMaxDynamicSharedMemorySize, 77824);
  (void)hipFuncSetAttribute((const void*)edge_conv_k<128, true>,
                            hipFuncAttributeMaxDynamicSharedMemorySize, 45056);
  (void)hipFuncSetAttribute((const void*)edge_conv_k<128, false>,
                            hipFuncAttributeMaxDynamicSharedMemorySize, 28672);

  auto pack = [&](const float* W, _Float16* dst, int Kreal, int Kpad, int F) {
    int total = Kpad * F;
    pack_w_k<<<(total + 255) / 256, 256, 0, stream>>>(W, dst, Kreal, Kpad, F);
  };
  pack(c01W1, p01_1, 6, 32, 128);   pack(c01W2, p01_2, 128, 128, 1024);
  pack(c11W1, p11_1, 6, 32, 128);   pack(c11W2, p11_2, 128, 128, 128);
  pack(c11W3, p11_3, 128, 128, 1024);
  pack(c10W1, p10_1, 6, 32, 128);   pack(c10W2, p10_2, 128, 128, 1024);
  pack(c00W1, p00_1, 6, 32, 256);   pack(c00W2, p00_2, 256, 256, 256);
  pack(c00W3, p00_3, 256, 256, 1024);
  pack(fo1W,  pfc1,  32, 32, 256);

  // step descriptors (V-cycle order): 0=conv_01, 1=conv_11, 2=conv_10, 3=conv_00
  const float* sea[4]  = { ead, eam + 131072 * 6, eau, eam };
  const int*   ssrc[4] = { eid, eim + 131072, eiu, eim };
  const int*   sdst[4] = { eid + 65536, eim + 163840 + 131072, eiu + 65536, eim + 163840 };
  const int    sE[4]   = { 65536, 32768, 65536, 131072 };

  // degree counts (fixed per conv)
  zero_i32_k<<<(4 * NND + 255) / 256, 256, 0, stream>>>(cnt, 4 * NND);
  for (int c = 0; c < 4; ++c)
    count_k<<<(sE[c] + 255) / 256, 256, 0, stream>>>(sdst[c], cnt + c * NND, sE[c]);

  // fc_in
  fc_in_k<<<(NND * WID + 255) / 256, 256, 0, stream>>>(x, fciW, fcib, h);

  // DEPTH=2 V-cycles
  for (int d = 0; d < 2; ++d) {
    for (int c = 0; c < 4; ++c) {
      zero_f32_k<<<(NND * WID + 255) / 256, 256, 0, stream>>>(s, NND * WID);
      int blocks = sE[c] / 64;
      if (c == 0) {
        edge_conv_k<128, false><<<blocks, 256, 28672, stream>>>(
            sea[c], ssrc[c], sdst[c], h, s, p01_1, c01b1, nullptr, nullptr, p01_2, c01b2);
      } else if (c == 1) {
        edge_conv_k<128, true><<<blocks, 256, 45056, stream>>>(
            sea[c], ssrc[c], sdst[c], h, s, p11_1, c11b1, p11_2, c11b2, p11_3, c11b3);
      } else if (c == 2) {
        edge_conv_k<128, false><<<blocks, 256, 28672, stream>>>(
            sea[c], ssrc[c], sdst[c], h, s, p10_1, c10b1, nullptr, nullptr, p10_2, c10b2);
      } else {
        edge_conv_k<256, true><<<blocks, 256, 77824, stream>>>(
            sea[c], ssrc[c], sdst[c], h, s, p00_1, c00b1, p00_2, c00b2, p00_3, c00b3);
      }
      finalize_k<<<(NND * WID + 255) / 256, 256, 0, stream>>>(h, s, cnt + c * NND);
    }
  }

  // output head
  fc_out1_k<<<32, 256, 0, stream>>>(h, pfc1, fo1b, t1);
  fc_out2_k<<<(NFINE + 255) / 256, 256, 0, stream>>>(t1, fo2W, fo2b, out);
}
